// VQEmbeddingMovingAverage_38328288149559
// MI455X (gfx1250) — matrix operand order repaired and checked by reading, not compiled, not run on hardware
//
#include <hip/hip_runtime.h>
#include <hip/hip_bf16.h>

typedef __attribute__((ext_vector_type(16))) _Float16 v16h;
typedef __attribute__((ext_vector_type(8)))  _Float16 v8h;
typedef __attribute__((ext_vector_type(8)))  float    v8f;

#define KCODES 1024
#define DDIM   512
#define NVEC   16384        // 64*256
#define DECAYF 0.99f
#define ONEMDF 0.01f

union AFrag { v16h v; v8h h[2]; };

// ---------------------------------------------------------------------------
// Kernel 1: codebook fp32 -> f16 copy + exact fp32 row norms (one wave per row)
// ---------------------------------------------------------------------------
__global__ void vq_prep_codebook(const float* __restrict__ cb,
                                 _Float16* __restrict__ cbh,
                                 float* __restrict__ cb_sqr) {
    const int lane = threadIdx.x & 31;
    const int row  = blockIdx.x * (blockDim.x >> 5) + (threadIdx.x >> 5);
    if (row >= KCODES) return;
    const float* r = cb + (size_t)row * DDIM;
    _Float16*    o = cbh + (size_t)row * DDIM;
    float acc = 0.f;
    #pragma unroll 4
    for (int i = lane; i < DDIM; i += 32) {
        float v = r[i];
        acc += v * v;
        o[i] = (_Float16)v;
    }
    #pragma unroll
    for (int off = 16; off > 0; off >>= 1) acc += __shfl_down(acc, off, 32);
    if (lane == 0) cb_sqr[row] = acc;
}

// ---------------------------------------------------------------------------
// Kernel 2: fused distance GEMM (WMMA f16 -> f32) + argmin over K
//   block = 256 threads (8 waves), 64 rows of X per block, all 1024 codes.
//   dist'(n,k) = ||c_k||^2 - 2 * <x_n, c_k>   (||x_n||^2 constant in argmin)
// ---------------------------------------------------------------------------
__global__ __launch_bounds__(256, 1)
void vq_argmin(const float* __restrict__ x,
               const _Float16* __restrict__ cbh,
               const float* __restrict__ cb_sqr,
               int* __restrict__ indices) {
    // padded row stride (520 halves = 1040B) to avoid LDS bank alignment
    __shared__ __align__(16) _Float16 As[64][520];
    __shared__ unsigned long long best[64];

    const int tid = threadIdx.x;
    const int rowBase = blockIdx.x * 64;

    // Stage X tile (fp32 global -> f16 LDS), coalesced
    for (int i = tid; i < 64 * DDIM; i += 256) {
        int r = i >> 9, c = i & 511;
        As[r][c] = (_Float16)x[(size_t)(rowBase + r) * DDIM + c];
    }
    if (tid < 64) best[tid] = ~0ull;
    __syncthreads();

    const int lane = tid & 31;
    const int wv   = tid >> 5;
    const int half = lane >> 4;   // which 16-lane half of the wave
    const int l16  = lane & 15;

    for (int it = 0; it < 8; ++it) {
        const int ct = wv + (it << 3);          // column tile 0..63
        const int colBase = ct << 4;
        // B fragment source: lane l16 = output column, contiguous 16 halves
        const _Float16* brow =
            cbh + (size_t)(colBase + l16) * DDIM + (half << 4);

        v8f acc0 = {}, acc1 = {}, acc2 = {}, acc3 = {};

        #pragma unroll 2
        for (int d0 = 0; d0 < DDIM; d0 += 32) {
            v16h b = *(const v16h*)(brow + d0);
            const int dlo = d0 + (half << 4);
            const int dhi = dlo + 8;
            AFrag a0, a1, a2, a3;
            a0.h[0] = *(const v8h*)&As[l16     ][dlo];
            a0.h[1] = *(const v8h*)&As[l16     ][dhi];
            a1.h[0] = *(const v8h*)&As[16 + l16][dlo];
            a1.h[1] = *(const v8h*)&As[16 + l16][dhi];
            a2.h[0] = *(const v8h*)&As[32 + l16][dlo];
            a2.h[1] = *(const v8h*)&As[32 + l16][dhi];
            a3.h[0] = *(const v8h*)&As[48 + l16][dlo];
            a3.h[1] = *(const v8h*)&As[48 + l16][dhi];
            acc0 = __builtin_amdgcn_wmma_f32_16x16x32_f16(false, a0.v, false, b,
                                                          (short)0, acc0, false, false);
            acc1 = __builtin_amdgcn_wmma_f32_16x16x32_f16(false, a1.v, false, b,
                                                          (short)0, acc1, false, false);
            acc2 = __builtin_amdgcn_wmma_f32_16x16x32_f16(false, a2.v, false, b,
                                                          (short)0, acc2, false, false);
            acc3 = __builtin_amdgcn_wmma_f32_16x16x32_f16(false, a3.v, false, b,
                                                          (short)0, acc3, false, false);
        }

        const int col = colBase + l16;
        const float cq = cb_sqr[col];
        v8f accs[4] = {acc0, acc1, acc2, acc3};
        #pragma unroll
        for (int rt = 0; rt < 4; ++rt) {
            #pragma unroll
            for (int e = 0; e < 8; ++e) {
                float dist = cq - 2.0f * accs[rt][e];
                unsigned u = __float_as_uint(dist);
                u = ((int)u < 0) ? ~u : (u | 0x80000000u);   // order-preserving map
                unsigned long long key =
                    ((unsigned long long)u << 32) | (unsigned)col;
                // C/D layout: VGPR e holds M = e + 8*half within the 16-row tile
                atomicMin(&best[(rt << 4) + (half << 3) + e], key);
            }
        }
    }
    __syncthreads();
    if (tid < 64)
        indices[rowBase + tid] = (int)(best[tid] & 0xFFFFFFFFull);
}

// ---------------------------------------------------------------------------
// Kernel 3a/3b: scatter counts and dw with global f32 atomics
// ---------------------------------------------------------------------------
__global__ void vq_scatter_counts(const int* __restrict__ idx,
                                  float* __restrict__ counts) {
    int n = blockIdx.x * blockDim.x + threadIdx.x;
    if (n < NVEC) atomicAdd(&counts[idx[n]], 1.0f);
}

__global__ void vq_scatter_dw(const float* __restrict__ x,
                              const int* __restrict__ idx,
                              float* __restrict__ dw) {
    size_t i = (size_t)blockIdx.x * blockDim.x + threadIdx.x;
    if (i >= (size_t)NVEC * DDIM) return;
    int row = (int)(i >> 9);
    int c   = (int)(i & 511);
    atomicAdd(&dw[(size_t)idx[row] * DDIM + c], x[i]);
}

// ---------------------------------------------------------------------------
// Kernel 4: new_ema_count = d*ema_count + (1-d)*counts
// ---------------------------------------------------------------------------
__global__ void vq_ema_count(const float* __restrict__ ema_count,
                             const float* __restrict__ counts,
                             float* __restrict__ out_cnt) {
    int k = blockIdx.x * blockDim.x + threadIdx.x;
    if (k < KCODES) out_cnt[k] = DECAYF * ema_count[k] + ONEMDF * counts[k];
}

// ---------------------------------------------------------------------------
// Kernel 5: new_ema_w and new_embedding = new_ema_w / new_ema_count
// ---------------------------------------------------------------------------
__global__ void vq_ema_w(const float* __restrict__ ema_w,
                         const float* __restrict__ dw,
                         const float* __restrict__ out_cnt,
                         float* __restrict__ out_emw,
                         float* __restrict__ out_emb) {
    size_t i = (size_t)blockIdx.x * blockDim.x + threadIdx.x;
    if (i >= (size_t)KCODES * DDIM) return;
    int k = (int)(i >> 9);
    float nw = DECAYF * ema_w[i] + ONEMDF * dw[i];
    out_emw[i] = nw;
    out_emb[i] = nw / out_cnt[k];
}

// ---------------------------------------------------------------------------
// Kernel 6: gather z_q_x (old codebook) and z_q_x_bar (new embedding)
// ---------------------------------------------------------------------------
__global__ void vq_gather(const float* __restrict__ codebook,
                          const float* __restrict__ emb,
                          const int* __restrict__ idx,
                          float* __restrict__ zq,
                          float* __restrict__ zqb) {
    size_t i = (size_t)blockIdx.x * blockDim.x + threadIdx.x;
    if (i >= (size_t)NVEC * DDIM) return;
    int row = (int)(i >> 9);
    int c   = (int)(i & 511);
    size_t src = (size_t)idx[row] * DDIM + c;
    zq[i]  = codebook[src];
    zqb[i] = emb[src];
}

// ---------------------------------------------------------------------------
extern "C" void kernel_launch(void* const* d_in, const int* in_sizes, int n_in,
                              void* d_out, int out_size, void* d_ws, size_t ws_size,
                              hipStream_t stream) {
    const float* z_e_x     = (const float*)d_in[0];   // [16384,512]
    const float* codebook  = (const float*)d_in[1];   // [1024,512]
    const float* ema_count = (const float*)d_in[2];   // [1024]
    const float* ema_w     = (const float*)d_in[3];   // [1024,512]

    // workspace layout (bytes)
    char* ws = (char*)d_ws;
    _Float16* cbh    = (_Float16*)(ws + 0);             // 1,048,576
    float*    cbsqr  = (float*)   (ws + 1048576);       //     4,096
    int*      idx    = (int*)     (ws + 1052672);       //    65,536
    float*    counts = (float*)   (ws + 1118208);       //     4,096
    float*    dw     = (float*)   (ws + 1122304);       // 2,097,152  (end 3,219,456)

    // output layout (floats, concatenated in return order)
    float* out   = (float*)d_out;
    float* o_zq  = out;                    // 8,388,608
    float* o_zqb = out + 8388608;          // 8,388,608
    float* o_emb = out + 16777216;         //   524,288
    float* o_cnt = out + 17301504;         //     1,024
    float* o_emw = out + 17302528;         //   524,288

    // zero counts + dw (contiguous region)
    hipMemsetAsync(ws + 1118208, 0, 4096 + 2097152, stream);

    // 1) codebook f16 + row norms: 1024 waves = 128 blocks x 256 thr
    vq_prep_codebook<<<128, 256, 0, stream>>>(codebook, cbh, cbsqr);

    // 2) fused WMMA distance + argmin: 256 blocks x 64 rows
    vq_argmin<<<NVEC / 64, 256, 0, stream>>>(z_e_x, cbh, cbsqr, idx);

    // 3) scatter
    vq_scatter_counts<<<(NVEC + 255) / 256, 256, 0, stream>>>(idx, counts);
    vq_scatter_dw<<<(NVEC * DDIM) / 256, 256, 0, stream>>>(z_e_x, idx, dw);

    // 4) EMA count
    vq_ema_count<<<(KCODES + 255) / 256, 256, 0, stream>>>(ema_count, counts, o_cnt);

    // 5) EMA w + embedding
    vq_ema_w<<<(KCODES * DDIM) / 256, 256, 0, stream>>>(ema_w, dw, o_cnt, o_emw, o_emb);

    // 6) gathers
    vq_gather<<<(NVEC * DDIM) / 256, 256, 0, stream>>>(codebook, o_emb, idx, o_zq, o_zqb);
}
